// LinearAttention_32384053412235
// MI455X (gfx1250) — matrix operand order repaired and checked by reading, not compiled, not run on hardware
//
#include <hip/hip_runtime.h>

#define NSP 4096   // spatial n = 64*64
#define TB  128    // block tile (M and N)
#define TK  32     // K chunk = WMMA K for bf16

typedef __attribute__((ext_vector_type(16))) __bf16 bf16x16;
typedef __attribute__((ext_vector_type(8)))  __bf16 bf16x8;
typedef __attribute__((ext_vector_type(8)))  float  fx8;
typedef __attribute__((ext_vector_type(4)))  float  fx4;

static __device__ __forceinline__ __bf16 f2bf(float f) {
  unsigned u = __builtin_bit_cast(unsigned, f);
  u += 0x7FFFu + ((u >> 16) & 1u);           // round-to-nearest-even
  unsigned short h = (unsigned short)(u >> 16);
  return __builtin_bit_cast(__bf16, h);
}

// C[b][m][n] = sum_k A[m][k] * B[b][k][n] (+ bias[m])
// A: M x K row-major (shared across batch), B: batch-strided K x NSP, C: batch-strided M x NSP
// Double-buffered LDS pipeline: global loads of tile k+1 overlap the WMMAs of tile k.
__global__ __launch_bounds__(256)
void gemm_wmma_bf16(const float* __restrict__ A,
                    const float* __restrict__ Bsrc,
                    float* __restrict__ Cdst,
                    const float* __restrict__ bias,
                    int K, size_t strideB, size_t strideC)
{
  // row stride 40 halfs = 80 bytes -> every fragment chunk is 16B aligned
  __shared__ __align__(16) __bf16 a_lds[2][TB * 40];
  __shared__ __align__(16) __bf16 b_lds[2][TB * 40];

  const int tid  = threadIdx.x;
  const int lane = tid & 31;
  const int wave = tid >> 5;
  const int wm = (wave >> 2) * 64;   // wave M offset within block (0,64)
  const int wn = (wave & 3) * 32;    // wave N offset within block (0,32,64,96)
  const int m0 = blockIdx.y * TB;
  const int n0 = blockIdx.x * TB;
  const float* Bp = Bsrc + (size_t)blockIdx.z * strideB;
  float*       Cp = Cdst + (size_t)blockIdx.z * strideC;

  // staging-address precompute (4 float4 chunks per thread per tile)
  int arow[4], ac4[4], brow[4], bn4[4];
  #pragma unroll
  for (int t = 0; t < 4; ++t) {
    int lin = tid + t * 256;          // 0..1023 float4 units
    arow[t] = lin >> 3;  ac4[t] = (lin & 7) << 2;    // A: [m 0..127][k4 0..28]
    brow[t] = lin >> 5;  bn4[t] = (lin & 31) << 2;   // B: [k 0..31][n4 0..124]
  }

  fx8 acc[4][2];
  #pragma unroll
  for (int mt = 0; mt < 4; ++mt)
    #pragma unroll
    for (int nt = 0; nt < 2; ++nt)
      acc[mt][nt] = (fx8){0.f,0.f,0.f,0.f,0.f,0.f,0.f,0.f};

  fx4 areg[4], breg[4];

  // ---- prologue: fetch tile k0=0 and stage into buffer 0 ----
  #pragma unroll
  for (int t = 0; t < 4; ++t) {
    areg[t] = *(const fx4*)(A  + (size_t)(m0 + arow[t]) * K + ac4[t]);
    breg[t] = *(const fx4*)(Bp + (size_t)brow[t] * NSP + n0 + bn4[t]);
  }
  #pragma unroll
  for (int t = 0; t < 4; ++t) {
    __bf16* d = &a_lds[0][arow[t] * 40 + ac4[t]];
    d[0] = f2bf(areg[t][0]); d[1] = f2bf(areg[t][1]);
    d[2] = f2bf(areg[t][2]); d[3] = f2bf(areg[t][3]);
    #pragma unroll
    for (int j = 0; j < 4; ++j)
      b_lds[0][(bn4[t] + j) * 40 + brow[t]] = f2bf(breg[t][j]);
  }
  __syncthreads();

  const int mrow = lane & 15;
  const int hi   = lane >> 4;

  int cur = 0;
  for (int k0 = 0; k0 < K; k0 += TK) {
    const bool has_next = (k0 + TK) < K;
    // ---- issue next tile's global loads early (hide behind WMMAs) ----
    if (has_next) {
      #pragma unroll
      for (int t = 0; t < 4; ++t) {
        areg[t] = *(const fx4*)(A  + (size_t)(m0 + arow[t]) * K + k0 + TK + ac4[t]);
        breg[t] = *(const fx4*)(Bp + (size_t)(k0 + TK + brow[t]) * NSP + n0 + bn4[t]);
      }
    }

    // ---- fragments from current buffer, per CDNA5 16-bit WMMA layouts ----
    bf16x16 af[4], bfr[2];
    #pragma unroll
    for (int mt = 0; mt < 4; ++mt) {
      // A 16x32: lanes 0-15: K 0-7 (v0-3) & 16-23 (v4-7); lanes 16-31: K 8-15 & 24-31
      const __bf16* base = &a_lds[cur][(wm + mt * 16 + mrow) * 40];
      bf16x8 lo = *(const bf16x8*)(base + hi * 8);
      bf16x8 up = *(const bf16x8*)(base + hi * 8 + 16);
      #pragma unroll
      for (int i = 0; i < 8; ++i) { af[mt][i] = lo[i]; af[mt][i + 8] = up[i]; }
    }
    #pragma unroll
    for (int nt = 0; nt < 2; ++nt) {
      // B 32x16: lanes 0-15: K 0-7 & 16-23; lanes 16-31: K 8-15 & 24-31 (matches A)
      const __bf16* base = &b_lds[cur][(wn + nt * 16 + mrow) * 40];
      bf16x8 lo = *(const bf16x8*)(base + hi * 8);
      bf16x8 up = *(const bf16x8*)(base + hi * 8 + 16);
      #pragma unroll
      for (int i = 0; i < 8; ++i) { bfr[nt][i] = lo[i]; bfr[nt][i + 8] = up[i]; }
    }
    #pragma unroll
    for (int mt = 0; mt < 4; ++mt)
      #pragma unroll
      for (int nt = 0; nt < 2; ++nt)
        acc[mt][nt] = __builtin_amdgcn_wmma_f32_16x16x32_bf16(
            false, af[mt], false, bfr[nt], (short)0, acc[mt][nt], false, false);

    // ---- convert & store staged tile into the alternate buffer ----
    if (has_next) {
      const int nxt = cur ^ 1;
      #pragma unroll
      for (int t = 0; t < 4; ++t) {
        __bf16* d = &a_lds[nxt][arow[t] * 40 + ac4[t]];
        d[0] = f2bf(areg[t][0]); d[1] = f2bf(areg[t][1]);
        d[2] = f2bf(areg[t][2]); d[3] = f2bf(areg[t][3]);
        #pragma unroll
        for (int j = 0; j < 4; ++j)
          b_lds[nxt][(bn4[t] + j) * 40 + brow[t]] = f2bf(breg[t][j]);
      }
    }
    __syncthreads();
    cur ^= 1;
  }

  // ---- epilogue: hoisted, branch-free bias add, then store D ----
  // D layout: VGPR r -> M = r + 8*(lane>=16), N = lane&15
  float badd[4][8];
  if (bias != nullptr) {
    #pragma unroll
    for (int mt = 0; mt < 4; ++mt) {
      int mbase = m0 + wm + mt * 16 + (lane >> 4) * 8;
      #pragma unroll
      for (int r = 0; r < 8; ++r) badd[mt][r] = bias[mbase + r];
    }
  } else {
    #pragma unroll
    for (int mt = 0; mt < 4; ++mt)
      #pragma unroll
      for (int r = 0; r < 8; ++r) badd[mt][r] = 0.f;
  }
  const int ncol = n0 + wn + (lane & 15);
  #pragma unroll
  for (int mt = 0; mt < 4; ++mt) {
    int mbase = m0 + wm + mt * 16 + (lane >> 4) * 8;
    #pragma unroll
    for (int nt = 0; nt < 2; ++nt) {
      #pragma unroll
      for (int r = 0; r < 8; ++r) {
        Cp[(size_t)(mbase + r) * NSP + ncol + nt * 16] = acc[mt][nt][r] + badd[mt][r];
      }
    }
  }
}

// One workgroup per (b, head). qkv layout: [b][768][4096]; rows 0-255 q, 256-511 k, 512-767 v.
// Writes attention output into the q rows (in place) for the final projection to consume.
__global__ __launch_bounds__(256)
void attn_kernel(float* __restrict__ qkv)
{
  const int b = blockIdx.y;
  const int h = blockIdx.x;
  float* q    = qkv + ((size_t)b * 768 + h * 32) * NSP;
  float* kmat = q + (size_t)256 * NSP;
  float* vmat = q + (size_t)512 * NSP;

  __shared__ float s_inv[32];
  __shared__ float s_ctx[32][33];
  __shared__ float s_kt[32][129];
  __shared__ float s_vt[32][129];

  const int tid  = threadIdx.x;
  const int lane = tid & 31;
  const int wave = tid >> 5;
  const float SCALE = 0.1767766952966369f;  // 32^-0.5

  // Phase 1: q softmax over d (axis=-2), fold in SCALE, in place.
  for (int n = tid; n < NSP; n += 256) {
    float col[32];
    float m = -3.402823466e38f;
    #pragma unroll
    for (int d = 0; d < 32; ++d) { col[d] = q[(size_t)d * NSP + n]; m = fmaxf(m, col[d]); }
    float s = 0.f;
    #pragma unroll
    for (int d = 0; d < 32; ++d) { col[d] = __expf(col[d] - m); s += col[d]; }
    float inv = SCALE / s;
    #pragma unroll
    for (int d = 0; d < 32; ++d) q[(size_t)d * NSP + n] = col[d] * inv;
  }

  // Phase 2: k row stats (axis=-1): unnormalized exp in place, 1/sum in LDS.
  #pragma unroll
  for (int rr = 0; rr < 4; ++rr) {
    int row = wave + rr * 8;
    float* kr = kmat + (size_t)row * NSP;
    float m = -3.402823466e38f;
    for (int n = lane; n < NSP; n += 32) m = fmaxf(m, kr[n]);
    #pragma unroll
    for (int off = 16; off > 0; off >>= 1) m = fmaxf(m, __shfl_xor(m, off, 32));
    float s = 0.f;
    for (int n = lane; n < NSP; n += 32) { float e = __expf(kr[n] - m); kr[n] = e; s += e; }
    #pragma unroll
    for (int off = 16; off > 0; off >>= 1) s += __shfl_xor(s, off, 32);
    if (lane == 0) s_inv[row] = 1.0f / s;
  }
  __syncthreads();

  // Phase 3: ctx[d][e] = (1/sum_d) * sum_n expk[d][n] * v[e][n]; LDS-tiled over n.
  float acc[4] = {0.f, 0.f, 0.f, 0.f};
  int di[4], ei[4];
  #pragma unroll
  for (int i = 0; i < 4; ++i) { int pi = tid + 256 * i; di[i] = pi >> 5; ei[i] = pi & 31; }
  for (int c0 = 0; c0 < NSP; c0 += 128) {
    #pragma unroll
    for (int t = 0; t < 16; ++t) {
      int lin = tid + t * 256;
      int r = lin >> 7, c = lin & 127;
      s_kt[r][c] = kmat[(size_t)r * NSP + c0 + c];
      s_vt[r][c] = vmat[(size_t)r * NSP + c0 + c];
    }
    __syncthreads();
    #pragma unroll
    for (int i = 0; i < 4; ++i) {
      float a = 0.f;
      #pragma unroll 8
      for (int j = 0; j < 128; ++j) a += s_kt[di[i]][j] * s_vt[ei[i]][j];
      acc[i] += a;
    }
    __syncthreads();
  }
  #pragma unroll
  for (int i = 0; i < 4; ++i) s_ctx[di[i]][ei[i]] = acc[i] * s_inv[di[i]];
  __syncthreads();

  // Phase 4: out[e][n] = sum_d ctx[d][e] * q[d][n]; overwrite q column-wise
  // (same thread owns the same columns as Phase 1 -> no cross-thread hazard).
  for (int n = tid; n < NSP; n += 256) {
    float qc[32], o[32];
    #pragma unroll
    for (int d = 0; d < 32; ++d) qc[d] = q[(size_t)d * NSP + n];
    #pragma unroll 4
    for (int e = 0; e < 32; ++e) {
      float t = 0.f;
      #pragma unroll
      for (int d = 0; d < 32; ++d) t += s_ctx[d][e] * qc[d];
      o[e] = t;
    }
    #pragma unroll
    for (int e = 0; e < 32; ++e) q[(size_t)e * NSP + n] = o[e];
  }
}

extern "C" void kernel_launch(void* const* d_in, const int* in_sizes, int n_in,
                              void* d_out, int out_size, void* d_ws, size_t ws_size,
                              hipStream_t stream) {
  const float* x     = (const float*)d_in[0];  // (16,256,64,64)
  const float* w_qkv = (const float*)d_in[1];  // (768,256)
  const float* w_out = (const float*)d_in[2];  // (256,256)
  const float* b_out = (const float*)d_in[3];  // (256,)
  float* out = (float*)d_out;                  // (16,256,64,64)
  float* qkv = (float*)d_ws;                   // 16*768*4096 f32 = 192 MiB

  // 1) qkv = w_qkv @ x   (per batch, 768 x 4096, K=256)
  gemm_wmma_bf16<<<dim3(NSP / TB, 768 / TB, 16), 256, 0, stream>>>(
      w_qkv, x, qkv, nullptr, 256, (size_t)256 * NSP, (size_t)768 * NSP);

  // 2) softmaxes + linear attention; writes result into the q rows of qkv
  attn_kernel<<<dim3(8, 16), 256, 0, stream>>>(qkv);

  // 3) out = w_out @ attn + b_out   (per batch, 256 x 4096, K=256)
  gemm_wmma_bf16<<<dim3(NSP / TB, 256 / TB, 16), 256, 0, stream>>>(
      w_out, qkv, out, b_out, 256, (size_t)768 * NSP, (size_t)256 * NSP);
}
